// BoTBlock_53223234732454
// MI455X (gfx1250) — matrix operand order repaired and checked by reading, not compiled, not run on hardware
//
#include <hip/hip_runtime.h>
#include <hip/hip_bf16.h>

// ---------------------------------------------------------------------------
// BoT block on gfx1250 (wave32, WMMA, TDM). All GEMMs: v_wmma_f32_16x16x32_bf16
// with fp32 accumulation. Flash attention avoids the 268MB attention matrix.
// A-tiles and attention operand streams are staged into LDS with the Tensor
// Data Mover (tensor_load_to_lds, TENSORcnt); LDS_FENCE makes the DMA's LDS
// writes visible to the compiler's memory model so ds_load reads materialize.
// ---------------------------------------------------------------------------

typedef __bf16 v16bf __attribute__((ext_vector_type(16)));
typedef __bf16 v8bf  __attribute__((ext_vector_type(8)));
typedef float  v8f   __attribute__((ext_vector_type(8)));
typedef unsigned v4u __attribute__((ext_vector_type(4)));
typedef int      v4i __attribute__((ext_vector_type(4)));
typedef int      v8i __attribute__((ext_vector_type(8)));

#define WMMA_BF16(A, B, C) \
  __builtin_amdgcn_wmma_f32_16x16x32_bf16(false, (A), false, (B), (short)0, (C), false, false)

#define CAT8(lo, hi) \
  __builtin_shufflevector((lo), (hi), 0, 1, 2, 3, 4, 5, 6, 7, 8, 9, 10, 11, 12, 13, 14, 15)

// Compiler cannot see that the TDM writes LDS: escape the pointer so reads of
// the shared array cannot be folded to undef and are ordered after the DMA.
#define LDS_FENCE(p) asm volatile("" : : "r"(p) : "memory")

#if __has_builtin(__builtin_amdgcn_tensor_load_to_lds)
#define HAVE_TDM 1
#else
#define HAVE_TDM 0
#endif

__device__ __forceinline__ __bf16 f2bf(float f) {
  unsigned u = __builtin_bit_cast(unsigned, f);
  unsigned r = u + 0x7FFFu + ((u >> 16) & 1u);   // round-to-nearest-even
  unsigned short s = (unsigned short)(r >> 16);
  return __builtin_bit_cast(__bf16, s);
}

__device__ __forceinline__ float rowmax16(float v) {
  v = fmaxf(v, __shfl_xor(v, 1));
  v = fmaxf(v, __shfl_xor(v, 2));
  v = fmaxf(v, __shfl_xor(v, 4));
  v = fmaxf(v, __shfl_xor(v, 8));
  return v;
}
__device__ __forceinline__ float rowsum16(float v) {
  v += __shfl_xor(v, 1);
  v += __shfl_xor(v, 2);
  v += __shfl_xor(v, 4);
  v += __shfl_xor(v, 8);
  return v;
}

#if HAVE_TDM
// TDM 2D tile load: bf16 elements (data_size=2B). D# per ISA 8.3/8.4:
// group0 = {count=1 | lds_addr | global_addr | type=2}, group1 = dims/strides.
__device__ __forceinline__ void tdm_load_2d(unsigned lds_byte_off, const void* gsrc,
                                            unsigned tile_x, unsigned tile_y,
                                            unsigned row_stride_elems,
                                            unsigned tensor_x, unsigned tensor_y) {
  unsigned long long ga = (unsigned long long)gsrc;
  v4u g0;
  g0[0] = 1u;                                             // count=1, user desc
  g0[1] = lds_byte_off;                                   // lds_addr (bytes)
  g0[2] = (unsigned)ga;                                   // global_addr[31:0]
  g0[3] = (unsigned)((ga >> 32) & 0x01FFFFFFu) | (2u << 30);  // ga[56:32] | type=2
  v8i g1;
  g1[0] = (int)(1u << 16);                                // data_size=1 -> 2 bytes
  g1[1] = (int)((tensor_x & 0xFFFFu) << 16);              // tensor_dim0[15:0]
  g1[2] = (int)(((tensor_x >> 16) & 0xFFFFu) | ((tensor_y & 0xFFFFu) << 16));
  g1[3] = (int)(((tensor_y >> 16) & 0xFFFFu) | ((tile_x & 0xFFFFu) << 16));   // tile_dim0
  g1[4] = (int)(tile_y & 0xFFFFu);                        // tile_dim1 (dim2=0)
  g1[5] = (int)row_stride_elems;                          // tensor_dim0_stride
  g1[6] = 0;
  g1[7] = 0;
  v4i z4 = {0, 0, 0, 0};
#if __clang_major__ >= 23
  v8i z8 = {0, 0, 0, 0, 0, 0, 0, 0};
  __builtin_amdgcn_tensor_load_to_lds(g0, g1, z4, z4, z8, 0);
#else
  __builtin_amdgcn_tensor_load_to_lds(g0, g1, z4, z4, 0);
#endif
}
#endif

// ---------------------------------------------------------------------------
// Weight fp32 -> bf16 (one-time, tiny).
// ---------------------------------------------------------------------------
__global__ __launch_bounds__(256) void cvt_k(const float* __restrict__ s,
                                             __bf16* __restrict__ d, int n) {
  int i = blockIdx.x * 256 + threadIdx.x;
  if (i < n) d[i] = f2bf(s[i]);
}

// ---------------------------------------------------------------------------
// x (B,C,HW) -> xbf[p, c] bf16 so conv1 A-tiles are contiguous rows.
// ---------------------------------------------------------------------------
__global__ __launch_bounds__(256) void xpose_k(const float* __restrict__ x,
                                               __bf16* __restrict__ xbf) {
  int idx = blockIdx.x * 256 + threadIdx.x;   // p*512 + c
  int c = idx & 511;
  int p = idx >> 9;
  int bb = p >> 10, hw = p & 1023;
  xbf[idx] = f2bf(x[((size_t)bb * 512 + c) * 1024 + hw]);
}

// ---------------------------------------------------------------------------
// GEMM1: y1[p,128] = xbf[p,c] @ w1bf^T   (M=16384, N=128, K=512)
// TDM stages the block-shared 16x512 A tile into LDS once (vs 4x redundant
// global loads); waves read A via ds_load_b128, stream B from global.
// ---------------------------------------------------------------------------
__global__ __launch_bounds__(128) void gemm_conv1_k(
    const __bf16* __restrict__ xbf, const __bf16* __restrict__ w1bf,
    float* __restrict__ y1) {
  __shared__ __bf16 sA[16 * 512];             // 16 KB, LDS offset 0
  int wv = threadIdx.x >> 5;
  int lane = threadIdx.x & 31, hf = lane >> 4, lid = lane & 15;
  int p0 = blockIdx.x * 16;
  int n0 = wv * 32;
#if HAVE_TDM
  if (wv == 0) {
    tdm_load_2d(0, xbf + (size_t)p0 * 512, 512, 16, 512, 512, 16384);
    __builtin_amdgcn_s_wait_tensorcnt((short)0);
  }
  LDS_FENCE(sA);
  __syncthreads();
#else
  for (int i = threadIdx.x; i < 16 * 512 / 8; i += 128)
    ((v8bf*)sA)[i] = ((const v8bf*)(xbf + (size_t)p0 * 512))[i];
  __syncthreads();
#endif
  const __bf16* arow = sA + lid * 512;
  v8f acc0 = {}, acc1 = {};
  for (int k0 = 0; k0 < 512; k0 += 32) {
    v8bf lo = *(const v8bf*)(arow + k0 + hf * 16);       // K = hf*16 + e
    v8bf hi = *(const v8bf*)(arow + k0 + hf * 16 + 8);   // K = hf*16 + 8 + e
    v16bf a = CAT8(lo, hi);
    v16bf b0 = *(const v16bf*)(w1bf + (size_t)(n0 + lid) * 512 + k0 + hf * 16);
    v16bf b1 = *(const v16bf*)(w1bf + (size_t)(n0 + 16 + lid) * 512 + k0 + hf * 16);
    acc0 = WMMA_BF16(a, b0, acc0);
    acc1 = WMMA_BF16(a, b1, acc1);
  }
#pragma unroll
  for (int i = 0; i < 8; ++i) {               // D: row = hf*8+i, col = lid
    int pr = p0 + hf * 8 + i;
    y1[(size_t)pr * 128 + n0 + lid]      = acc0[i];
    y1[(size_t)pr * 128 + n0 + 16 + lid] = acc1[i];
  }
}

// ---------------------------------------------------------------------------
// Train-mode BN stats folded to affine: v*scale + shift.
// ---------------------------------------------------------------------------
__global__ __launch_bounds__(256) void bn_stats_k(
    const float* __restrict__ y, const float* __restrict__ g,
    const float* __restrict__ be, float* __restrict__ scale,
    float* __restrict__ shift, int C) {
  __shared__ float ss[256], sq[256];
  int c = blockIdx.x;
  float s = 0.f, q = 0.f;
  for (int p = threadIdx.x; p < 16384; p += 256) {
    float v = y[(size_t)p * C + c];
    s += v; q += v * v;
  }
  ss[threadIdx.x] = s; sq[threadIdx.x] = q;
  __syncthreads();
  for (int o = 128; o > 0; o >>= 1) {
    if ((int)threadIdx.x < o) {
      ss[threadIdx.x] += ss[threadIdx.x + o];
      sq[threadIdx.x] += sq[threadIdx.x + o];
    }
    __syncthreads();
  }
  if (threadIdx.x == 0) {
    float m = ss[0] * (1.0f / 16384.0f);
    float v = sq[0] * (1.0f / 16384.0f) - m * m;
    float sc = rsqrtf(v + 1e-5f) * g[c];
    scale[c] = sc;
    shift[c] = be[c] - m * sc;
  }
}

// ---------------------------------------------------------------------------
// BN1 + ReLU + bf16 convert (keeps gemm_qkv a pure bf16 GEMM).
// ---------------------------------------------------------------------------
__global__ __launch_bounds__(256) void bn1_apply_k(
    const float* __restrict__ y1, const float* __restrict__ scale1,
    const float* __restrict__ shift1, __bf16* __restrict__ y1bf) {
  int idx = blockIdx.x * 256 + threadIdx.x;   // 16384*128
  int c = idx & 127;
  float v = fmaxf(fmaf(y1[idx], scale1[c], shift1[c]), 0.f);
  y1bf[idx] = f2bf(v);
}

// ---------------------------------------------------------------------------
// GEMM2: qkv[b,o,n] = y1bf[p,c] @ wqbf^T   (M=16384, N=384, K=128)
// TDM stages the 16x128 A tile (4 KB) per block.
// ---------------------------------------------------------------------------
__global__ __launch_bounds__(128) void gemm_qkv_k(
    const __bf16* __restrict__ y1bf, const __bf16* __restrict__ wqbf,
    float* __restrict__ qkvb) {
  __shared__ __bf16 sA[16 * 128];
  int wv = threadIdx.x >> 5;
  int lane = threadIdx.x & 31, hf = lane >> 4, lid = lane & 15;
  int n0 = blockIdx.y * 128 + wv * 32;
  int p0 = blockIdx.x * 16;
#if HAVE_TDM
  if (wv == 0) {
    tdm_load_2d(0, y1bf + (size_t)p0 * 128, 128, 16, 128, 128, 16384);
    __builtin_amdgcn_s_wait_tensorcnt((short)0);
  }
  LDS_FENCE(sA);
  __syncthreads();
#else
  for (int i = threadIdx.x; i < 16 * 128 / 8; i += 128)
    ((v8bf*)sA)[i] = ((const v8bf*)(y1bf + (size_t)p0 * 128))[i];
  __syncthreads();
#endif
  const __bf16* arow = sA + lid * 128;
  v8f acc0 = {}, acc1 = {};
#pragma unroll
  for (int k0 = 0; k0 < 128; k0 += 32) {
    v8bf lo = *(const v8bf*)(arow + k0 + hf * 16);
    v8bf hi = *(const v8bf*)(arow + k0 + hf * 16 + 8);
    v16bf a = CAT8(lo, hi);
    v16bf b0 = *(const v16bf*)(wqbf + (size_t)(n0 + lid) * 128 + k0 + hf * 16);
    v16bf b1 = *(const v16bf*)(wqbf + (size_t)(n0 + 16 + lid) * 128 + k0 + hf * 16);
    acc0 = WMMA_BF16(a, b0, acc0);
    acc1 = WMMA_BF16(a, b1, acc1);
  }
#pragma unroll
  for (int i = 0; i < 8; ++i) {
    int pr = p0 + hf * 8 + i;
    int bb = pr >> 10, nn = pr & 1023;
    qkvb[((size_t)bb * 384 + n0 + lid) * 1024 + nn]      = acc0[i];
    qkvb[((size_t)bb * 384 + n0 + 16 + lid) * 1024 + nn] = acc1[i];
  }
}

// ---------------------------------------------------------------------------
// Stage q / (k + r) / v into exact WMMA fragment order (contiguous per tile):
//   qs  [bh][ntile][lane][16]        A-frag (row n, K = d)
//   kks [bh][mtile][lane][16]        B-frag (col m, K = d), rel-pos folded in
//   vs  [bh][mchunk][slot][lane][16] B-frag (col d, K = m)
// ---------------------------------------------------------------------------
__global__ __launch_bounds__(256) void prep_frag_k(
    const float* __restrict__ qkvb, const float* __restrict__ rw,
    const float* __restrict__ rh, __bf16* __restrict__ qs,
    __bf16* __restrict__ kks, __bf16* __restrict__ vs) {
  int t = blockIdx.x * 256 + threadIdx.x;    // 3 * 131072
  int kind = t >> 17;
  int r = t & 131071;
  int lane = r & 31, hf = lane >> 4, lid = lane & 15;
  int bh = r >> 11;
  int b = bh >> 2, h = bh & 3;
  size_t qb = (size_t)b * 384 * 1024;
  v16bf frag;
  if (kind == 0) {
    int tile = (r >> 5) & 63;
    int n = tile * 16 + lid;
#pragma unroll
    for (int e = 0; e < 8; ++e) {
      frag[e]     = f2bf(qkvb[qb + (size_t)(h * 32 + hf * 16 + e) * 1024 + n]);
      frag[8 + e] = f2bf(qkvb[qb + (size_t)(h * 32 + hf * 16 + 8 + e) * 1024 + n]);
    }
    *(v16bf*)(qs + (size_t)r * 16) = frag;
  } else if (kind == 1) {
    int tile = (r >> 5) & 63;
    int m = tile * 16 + lid;
#pragma unroll
    for (int e = 0; e < 16; ++e) {
      int d = hf * 16 + e;
      float rv = rw[(h * 32 + d) * 32 + (m & 31)] + rh[(h * 32 + d) * 32 + (m >> 5)];
      frag[e] = f2bf(qkvb[qb + (size_t)(128 + h * 32 + d) * 1024 + m] + rv);
    }
    *(v16bf*)(kks + (size_t)r * 16) = frag;
  } else {
    int slot = (r >> 5) & 1;
    int mc = (r >> 6) & 31;
    int d = slot * 16 + lid;
    const float* vrow = qkvb + qb + (size_t)(256 + h * 32 + d) * 1024 + mc * 32 + hf * 16;
#pragma unroll
    for (int e = 0; e < 16; ++e) frag[e] = f2bf(vrow[e]);
    *(v16bf*)(vs + (size_t)r * 16) = frag;
  }
}

// ---------------------------------------------------------------------------
// Flash attention, TDM double-buffered: per wave, chunk mc is a contiguous
// 2KB kk block + 2KB v block. Prefetch chunk mc+2 with tensor_load_to_lds
// while computing on mc; s_wait_tensorcnt<=2 gates buffer reuse (TDM completes
// in-order per wave); s_wait_dscnt 0 orders DS reads vs the overwriting DMA.
// ---------------------------------------------------------------------------
__global__ __launch_bounds__(128) void attention_k(
    const __bf16* __restrict__ qs, const __bf16* __restrict__ kks,
    const __bf16* __restrict__ vs, __bf16* __restrict__ y2bf) {
  // [0,32768): per-wave TDM buffers (wv*8192 + phase*4096 + {0:kk, 2048:v})
  // [32768,...): softmax bounce tiles, 4 x 16 x 33 floats
  __shared__ char smem[32768 + 4 * 16 * 33 * 4];
  int wv = threadIdx.x >> 5;
  int lane = threadIdx.x & 31, hf = lane >> 4, lid = lane & 15;
  int wid = blockIdx.x * 4 + wv;
  int b = wid >> 8, h = (wid >> 6) & 3, nt = wid & 63;
  int bh = b * 4 + h;
  const float scale = 0.17677669529663687f;   // 1/sqrt(32)
  float* sP = (float*)(smem + 32768) + wv * 528;        // [16][33]

  v16bf aq = *(const v16bf*)(qs + ((size_t)wid * 32 + lane) * 16);
  const __bf16* kkc = kks + (size_t)bh * 32768;         // + mc*1024, contiguous
  const __bf16* vc  = vs  + (size_t)bh * 32768;         // + mc*1024, contiguous

  float rmax[8], rsum[8];
#pragma unroll
  for (int i = 0; i < 8; ++i) { rmax[i] = -3.0e38f; rsum[i] = 0.f; }
  v8f acc0 = {}, acc1 = {};

#if HAVE_TDM
  unsigned lbase = (unsigned)wv * 8192u;
  // prologue: fetch chunks 0 and 1
  tdm_load_2d(lbase,            kkc,        1024, 1, 1024, 1024, 1);
  tdm_load_2d(lbase + 2048,     vc,         1024, 1, 1024, 1024, 1);
  tdm_load_2d(lbase + 4096,     kkc + 1024, 1024, 1, 1024, 1024, 1);
  tdm_load_2d(lbase + 6144,     vc + 1024,  1024, 1, 1024, 1024, 1);
  LDS_FENCE(smem);
#endif

  for (int mc = 0; mc < 32; ++mc) {
#if HAVE_TDM
    __builtin_amdgcn_s_wait_tensorcnt((short)2);        // chunk mc landed
    LDS_FENCE(smem);
    const __bf16* kbuf = (const __bf16*)(smem + lbase + (unsigned)(mc & 1) * 4096u);
    const __bf16* vbuf = kbuf + 1024;
#else
    const __bf16* kbuf = kkc + (size_t)mc * 1024;
    const __bf16* vbuf = vc + (size_t)mc * 1024;
#endif
    v16bf bk0 = *(const v16bf*)(kbuf + lane * 16);
    v16bf bk1 = *(const v16bf*)(kbuf + 512 + lane * 16);
    v8f s0 = {}, s1 = {};
    s0 = WMMA_BF16(aq, bk0, s0);
    s1 = WMMA_BF16(aq, bk1, s1);

    // online softmax: D rows = hf*8+i, cols spread across 16 lanes
#pragma unroll
    for (int i = 0; i < 8; ++i) {
      float e0 = s0[i] * scale, e1 = s1[i] * scale;
      float tm = rowmax16(fmaxf(e0, e1));
      float nm = fmaxf(rmax[i], tm);
      float corr = __expf(rmax[i] - nm);
      float q0 = __expf(e0 - nm), q1 = __expf(e1 - nm);
      float ts = rowsum16(q0 + q1);
      rsum[i] = rsum[i] * corr + ts;
      rmax[i] = nm;
      acc0[i] *= corr; acc1[i] *= corr;
      sP[(hf * 8 + i) * 33 + lid]      = q0;
      sP[(hf * 8 + i) * 33 + 16 + lid] = q1;
    }
    asm volatile("s_wait_dscnt 0x0" ::: "memory");

    v16bf ap;
#pragma unroll
    for (int e = 0; e < 8; ++e) {
      ap[e]     = f2bf(sP[lid * 33 + hf * 16 + e]);
      ap[8 + e] = f2bf(sP[lid * 33 + hf * 16 + 8 + e]);
    }
    v16bf bv0 = *(const v16bf*)(vbuf + lane * 16);
    v16bf bv1 = *(const v16bf*)(vbuf + 512 + lane * 16);
    acc0 = WMMA_BF16(ap, bv0, acc0);
    acc1 = WMMA_BF16(ap, bv1, acc1);
    asm volatile("s_wait_dscnt 0x0" ::: "memory");      // DS reads done
#if HAVE_TDM
    if (mc + 2 < 32) {                                  // prefetch into freed phase
      unsigned lph = lbase + (unsigned)(mc & 1) * 4096u;
      tdm_load_2d(lph,        kkc + (size_t)(mc + 2) * 1024, 1024, 1, 1024, 1024, 1);
      tdm_load_2d(lph + 2048, vc  + (size_t)(mc + 2) * 1024, 1024, 1, 1024, 1024, 1);
      LDS_FENCE(smem);
    }
#endif
  }
#if HAVE_TDM
  __builtin_amdgcn_s_wait_tensorcnt((short)0);
#endif

#pragma unroll
  for (int i = 0; i < 8; ++i) {
    float inv = 1.0f / rsum[i];
    int nn = nt * 16 + hf * 8 + i;
    size_t p = (size_t)b * 1024 + nn;
    y2bf[p * 128 + h * 32 + lid]      = f2bf(acc0[i] * inv);
    y2bf[p * 128 + h * 32 + 16 + lid] = f2bf(acc1[i] * inv);
  }
}

// ---------------------------------------------------------------------------
// GEMM3: z[p,512] = y2bf[p,ch] @ w2bf^T   (M=16384, N=512, K=128)
// ---------------------------------------------------------------------------
__global__ __launch_bounds__(128) void gemm_conv2_k(
    const __bf16* __restrict__ y2bf, const __bf16* __restrict__ w2bf,
    float* __restrict__ z) {
  __shared__ __bf16 sA[16 * 128];
  int wv = threadIdx.x >> 5;
  int lane = threadIdx.x & 31, hf = lane >> 4, lid = lane & 15;
  int n0 = blockIdx.y * 128 + wv * 32;
  int p0 = blockIdx.x * 16;
#if HAVE_TDM
  if (wv == 0) {
    tdm_load_2d(0, y2bf + (size_t)p0 * 128, 128, 16, 128, 128, 16384);
    __builtin_amdgcn_s_wait_tensorcnt((short)0);
  }
  LDS_FENCE(sA);
  __syncthreads();
#else
  for (int i = threadIdx.x; i < 16 * 128 / 8; i += 128)
    ((v8bf*)sA)[i] = ((const v8bf*)(y2bf + (size_t)p0 * 128))[i];
  __syncthreads();
#endif
  const __bf16* arow = sA + lid * 128;
  v8f acc0 = {}, acc1 = {};
#pragma unroll
  for (int k0 = 0; k0 < 128; k0 += 32) {
    v8bf lo = *(const v8bf*)(arow + k0 + hf * 16);
    v8bf hi = *(const v8bf*)(arow + k0 + hf * 16 + 8);
    v16bf a = CAT8(lo, hi);
    v16bf b0 = *(const v16bf*)(w2bf + (size_t)(n0 + lid) * 128 + k0 + hf * 16);
    v16bf b1 = *(const v16bf*)(w2bf + (size_t)(n0 + 16 + lid) * 128 + k0 + hf * 16);
    acc0 = WMMA_BF16(a, b0, acc0);
    acc1 = WMMA_BF16(a, b1, acc1);
  }
#pragma unroll
  for (int i = 0; i < 8; ++i) {
    int pr = p0 + hf * 8 + i;
    z[(size_t)pr * 512 + n0 + lid]      = acc0[i];
    z[(size_t)pr * 512 + n0 + 16 + lid] = acc1[i];
  }
}

// ---------------------------------------------------------------------------
// Epilogue: bn2 (affine) + residual + relu -> (B, C, H, W).
// ---------------------------------------------------------------------------
__global__ __launch_bounds__(256) void bn2_residual_k(
    const float* __restrict__ z, const float* __restrict__ x,
    const float* __restrict__ scale2, const float* __restrict__ shift2,
    float* __restrict__ out) {
  int idx = blockIdx.x * 256 + threadIdx.x;
  int n = idx & 1023;
  int c = (idx >> 10) & 511;
  int b = idx >> 19;
  float v = z[((size_t)(b * 1024 + n)) * 512 + c];
  v = fmaf(v, scale2[c], shift2[c]) + x[idx];
  out[idx] = fmaxf(v, 0.f);
}

// ---------------------------------------------------------------------------
extern "C" void kernel_launch(void* const* d_in, const int* in_sizes, int n_in,
                              void* d_out, int out_size, void* d_ws, size_t ws_size,
                              hipStream_t stream) {
  const float* x   = (const float*)d_in[0];
  const float* w1  = (const float*)d_in[1];
  const float* g1  = (const float*)d_in[2];
  const float* be1 = (const float*)d_in[3];
  const float* wq  = (const float*)d_in[4];
  const float* rw  = (const float*)d_in[5];
  const float* rh  = (const float*)d_in[6];
  const float* w2  = (const float*)d_in[7];
  const float* g2  = (const float*)d_in[8];
  const float* be2 = (const float*)d_in[9];
  float* out = (float*)d_out;

  // Workspace (bytes). z (32MB @0) overlaps xbf/y1/y1bf/front of qkvb — all
  // dead before gemm_conv2. Everything stays L2-resident (<192MB).
  char* ws = (char*)d_ws;
  __bf16* xbf    = (__bf16*)(ws);                    // 16 MB: [p,512]
  float*  y1     = (float*)(ws + 16777216);          //  8 MB: [p,128]
  __bf16* y1bf   = (__bf16*)(ws + 25165824);         //  4 MB
  float*  qkvb   = (float*)(ws + 29360128);          // 24 MB: [b,384,1024]
  __bf16* qs     = (__bf16*)(ws + 54525952);         //  4 MB staged A-frags
  __bf16* kks    = (__bf16*)(ws + 58720256);         //  4 MB staged B-frags
  __bf16* vsb    = (__bf16*)(ws + 62914560);         //  4 MB staged B-frags
  __bf16* y2bf   = (__bf16*)(ws + 67108864);         //  4 MB: [p,128]
  float*  z      = (float*)(ws);                     // 32 MB: [p,512] (overlap)
  __bf16* w1bf   = (__bf16*)(ws + 71303168);         // 128 KB
  __bf16* wqbf   = (__bf16*)(ws + 71434240);         //  96 KB
  __bf16* w2bf   = (__bf16*)(ws + 71532544);         // 128 KB
  float*  scale1 = (float*)(ws + 71663616);          // 128
  float*  shift1 = scale1 + 128;
  float*  scale2 = shift1 + 128;                     // 512
  float*  shift2 = scale2 + 512;

  cvt_k<<<256, 256, 0, stream>>>(w1, w1bf, 65536);
  cvt_k<<<192, 256, 0, stream>>>(wq, wqbf, 49152);
  cvt_k<<<256, 256, 0, stream>>>(w2, w2bf, 65536);
  xpose_k<<<32768, 256, 0, stream>>>(x, xbf);
  gemm_conv1_k<<<1024, 128, 0, stream>>>(xbf, w1bf, y1);
  bn_stats_k<<<128, 256, 0, stream>>>(y1, g1, be1, scale1, shift1, 128);
  bn1_apply_k<<<8192, 256, 0, stream>>>(y1, scale1, shift1, y1bf);
  gemm_qkv_k<<<dim3(1024, 3), 128, 0, stream>>>(y1bf, wqbf, qkvb);
  prep_frag_k<<<1536, 256, 0, stream>>>(qkvb, rw, rh, qs, kks, vsb);
  attention_k<<<1024, 128, 0, stream>>>(qs, kks, vsb, y2bf);
  gemm_conv2_k<<<dim3(1024, 4), 128, 0, stream>>>(y2bf, w2bf, z);
  bn_stats_k<<<512, 256, 0, stream>>>(z, g2, be2, scale2, shift2, 512);
  bn2_residual_k<<<32768, 256, 0, stream>>>(z, x, scale2, shift2, out);
}
